// DecoderGRUWithAttention_41712722379266
// MI455X (gfx1250) — matrix operand order repaired and checked by reading, not compiled, not run on hardware
//
#include <hip/hip_runtime.h>
#include <math.h>

#define B_   64
#define S_   2048
#define IN_  256
#define H_   512
#define E_   256
#define OUT_ 32000
#define KP_  (H_ + 8)          // padded LDS k-stride (bank spread)
#define NCHUNK_ (S_ / 128)     // 16 score/context chunks per batch row

typedef __attribute__((ext_vector_type(16))) __bf16 v16bf;
typedef __attribute__((ext_vector_type(8)))  __bf16 v8bf;
typedef __attribute__((ext_vector_type(2)))  __bf16 v2bf;
typedef __attribute__((ext_vector_type(8)))  float  v8f;

__device__ __forceinline__ __bf16 f2bf(float f) {
    unsigned u = __builtin_bit_cast(unsigned, f);
    unsigned r = u + 0x7FFFu + ((u >> 16) & 1u);   // round-to-nearest-even
    unsigned short h = (unsigned short)(r >> 16);
    return __builtin_bit_cast(__bf16, h);
}
// two f32 -> packed 2 x bf16; V_CVT_PK_BF16_F32 when available
__device__ __forceinline__ v2bf cvt2(float a, float b) {
#if __has_builtin(__builtin_amdgcn_cvt_pk_bf16_f32)
    return __builtin_amdgcn_cvt_pk_bf16_f32(a, b);
#else
    v2bf o; o[0] = f2bf(a); o[1] = f2bf(b); return o;
#endif
}
__device__ __forceinline__ float bf2f(__bf16 h) {
    unsigned short s = __builtin_bit_cast(unsigned short, h);
    unsigned u = ((unsigned)s) << 16;
    return __builtin_bit_cast(float, u);
}
__device__ __forceinline__ float fast_tanh(float x) {
#if __has_builtin(__builtin_amdgcn_tanhf)
    return __builtin_amdgcn_tanhf(x);      // V_TANH_F32 (CDNA5 trans op)
#else
    const float e = __expf(2.f * x);
    return 1.f - 2.f / (e + 1.f);
#endif
}
__device__ __forceinline__ float fast_sigmoid(float x) {
    return 1.f / (1.f + __expf(-x));
}

// A-fragment (16x32 bf16) from row-major f32 (ISA 7.12.2: lanes 0-15 hold
// K +0..7 / +16..23, lanes 16-31 hold K +8..15 / +24..31).
// Reads via float4 (all A sources are 16B aligned, k offsets multiple of 8).
__device__ __forceinline__ v16bf load_a_frag_f32(const float* A, int lda, int row0,
                                                 int k0, int lane) {
    const int r  = row0 + (lane & 15);
    const int kb = k0 + ((lane >> 4) << 3);
    const float4* p = (const float4*)(A + (long)r * lda + kb);
    const float4 f0 = p[0], f1 = p[1];   // K kb   .. kb+7
    const float4 f2 = p[4], f3 = p[5];   // K kb+16.. kb+23
    union { v16bf v; v2bf e[8]; } u;
    u.e[0] = cvt2(f0.x, f0.y); u.e[1] = cvt2(f0.z, f0.w);
    u.e[2] = cvt2(f1.x, f1.y); u.e[3] = cvt2(f1.z, f1.w);
    u.e[4] = cvt2(f2.x, f2.y); u.e[5] = cvt2(f2.z, f2.w);
    u.e[6] = cvt2(f3.x, f3.y); u.e[7] = cvt2(f3.z, f3.w);
    return u.v;
}

// A-fragment over the virtual concat(x[:,0:256], context[:,0:512]) operand.
// k-steps of 32 never straddle the 256 boundary.
__device__ __forceinline__ v16bf load_a_xc(const float* x, const float* ctx,
                                           int row0, int k0, int lane) {
    if (k0 < IN_) return load_a_frag_f32(x, IN_, row0, k0, lane);
    return load_a_frag_f32(ctx, H_, row0, k0 - IN_, lane);
}

// B-fragment (32x16 bf16) from row-major f32 B[k, n]: lane = column, halves = K.
__device__ __forceinline__ v16bf load_b_frag_f32(const float* Bm, int ldb, int k0,
                                                 int n0, int lane) {
    const int n  = n0 + (lane & 15);
    const int kb = k0 + ((lane >> 4) << 3);
    const float* p = Bm + (long)kb * ldb + n;
    union { v16bf v; v2bf e[8]; } u;
#pragma unroll
    for (int j = 0; j < 16; j += 2)
        u.e[j >> 1] = cvt2(p[(long)(j + (j & 8)) * ldb], p[(long)(j + 1 + (j & 8)) * ldb]);
    return u.v;
}

// ---------------------------------------------------------------------------
// pre[b,e] = hidden[b,:] @ W2[:,e] + W1_b[e]   (tiny, scalar f32)
// ---------------------------------------------------------------------------
__global__ void k_pre(const float* __restrict__ hidden, const float* __restrict__ W2,
                      const float* __restrict__ W1b, float* __restrict__ pre) {
    const int idx = blockIdx.x * blockDim.x + threadIdx.x;
    const int b = idx / E_, e = idx % E_;
    const float* hrow = hidden + b * H_;
    float acc = W1b[e];
    for (int h = 0; h < H_; ++h) acc += hrow[h] * W2[h * E_ + e];
    pre[idx] = acc;
}

// ---------------------------------------------------------------------------
// Pack W1 (f32 [H,E]) into bf16 WMMA B-fragment order:
//   frag(e_tile, k_step): 32 lanes x 16 bf16 contiguous per lane.
// ---------------------------------------------------------------------------
__global__ void k_pack_w1(const float* __restrict__ W1, __bf16* __restrict__ W1p) {
    const int idx  = blockIdx.x * blockDim.x + threadIdx.x;  // H_*E_ = 131072
    const int j    = idx & 15;
    const int lane = (idx >> 4) & 31;
    const int ks   = (idx >> 9) & 15;
    const int e    = idx >> 13;
    const int n    = e * 16 + (lane & 15);
    const int k    = ks * 32 + ((lane >> 4) << 3) + j + (j & 8);
    W1p[idx] = f2bf(W1[k * E_ + n]);
}

// ---------------------------------------------------------------------------
// Fused attention chunk kernel (flash-style): for 128 s-rows of one batch row,
//   - stage enc tile to LDS (bf16, padded stride, packed cvt + b128 loads)
//   - bf16 WMMA:  energy = encT @ W1 ; score = v . tanh(energy + pre)
//   - raw scores -> d_out attn slot (normalized later)
//   - local softmax (m_c, l_c) + partial context from the LDS tile
// encoder_outputs (268 MB) touches HBM exactly once.
// ---------------------------------------------------------------------------
__global__ void k_attn_chunk(const float* __restrict__ enc, const __bf16* __restrict__ W1p,
                             const float* __restrict__ pre, const float* __restrict__ v_w,
                             float* __restrict__ raw_scores, float* __restrict__ partM,
                             float* __restrict__ partL, float* __restrict__ partCtx) {
    extern __shared__ char smem[];
    __bf16* As  = (__bf16*)smem;                                   // 128 x KP_
    float* sArr = (float*)(smem + (size_t)128 * KP_ * sizeof(__bf16));
    float* wArr = sArr + 128;
    float* redB = wArr + 128;

    const int b     = blockIdx.y;
    const int chunk = blockIdx.x;
    const int row0  = chunk * 128;
    const int t     = threadIdx.x;
    const int lane  = t & 31;
    const int wave  = t >> 5;

    // ---- stage enc tile: global f32 (b128) -> LDS bf16 (packed cvt)
    {
        const int r  = t >> 1;
        const int k0 = (t & 1) * 256;
        const float4* s4 = (const float4*)(enc + ((long)b * S_ + row0 + r) * H_ + k0);
        __bf16* dst = As + r * KP_ + k0;
#pragma unroll 4
        for (int j = 0; j < 64; j += 2) {          // 64 float4 = 256 floats
            const float4 f0 = s4[j], f1 = s4[j + 1];
            union { v8bf v; v2bf e[4]; } u;
            u.e[0] = cvt2(f0.x, f0.y); u.e[1] = cvt2(f0.z, f0.w);
            u.e[2] = cvt2(f1.x, f1.y); u.e[3] = cvt2(f1.z, f1.w);
            *(v8bf*)(dst + j * 4) = u.v;
        }
    }
    __syncthreads();

    // ---- WMMA scores: per wave a 16-row s-subtile; 2 groups of 8 e-tiles so
    //      each A fragment feeds 8 unrolled WMMAs.
    const int rw  = wave * 16 + (lane & 15);
    const int kba = (lane >> 4) << 3;
    float sacc[8];
#pragma unroll
    for (int i = 0; i < 8; ++i) sacc[i] = 0.f;

    for (int eg = 0; eg < 2; ++eg) {
        v8f zero = {};
        v8f acc[8] = {zero, zero, zero, zero, zero, zero, zero, zero};
        for (int k0 = 0; k0 < H_; k0 += 32) {
            const __bf16* pa = As + rw * KP_ + k0 + kba;
            v8bf lo = *(const v8bf*)pa;
            v8bf hi = *(const v8bf*)(pa + 16);
            v16bf a = __builtin_shufflevector(lo, hi, 0, 1, 2, 3, 4, 5, 6, 7,
                                              8, 9, 10, 11, 12, 13, 14, 15);
            const int ks = k0 >> 5;
#pragma unroll
            for (int et = 0; et < 8; ++et) {
                const int e = eg * 8 + et;
                v16bf bb = *(const v16bf*)(W1p + ((((e * 16) + ks) * 32 + lane) << 4));
                acc[et] = __builtin_amdgcn_wmma_f32_16x16x32_bf16(
                    false, a, false, bb, (short)0, acc[et], false, false);
            }
        }
#pragma unroll
        for (int et = 0; et < 8; ++et) {
            const int   col = (eg * 8 + et) * 16 + (lane & 15);
            const float pv  = pre[b * E_ + col];
            const float vv  = v_w[col];
#pragma unroll
            for (int i = 0; i < 8; ++i) sacc[i] += fast_tanh(acc[et][i] + pv) * vv;
        }
    }

    // reduce over the 16 columns held per half-wave
#pragma unroll
    for (int i = 0; i < 8; ++i) {
        float s = sacc[i];
        s += __shfl_xor(s, 1, 16);
        s += __shfl_xor(s, 2, 16);
        s += __shfl_xor(s, 4, 16);
        s += __shfl_xor(s, 8, 16);
        sacc[i] = s;
    }
    if ((lane & 15) == 0) {
        const int rloc = wave * 16 + ((lane >> 4) << 3);
#pragma unroll
        for (int i = 0; i < 8; ++i) {
            sArr[rloc + i] = sacc[i];
            raw_scores[(long)b * S_ + row0 + rloc + i] = sacc[i];
        }
    }
    __syncthreads();

    // ---- local softmax over the 128 chunk rows
    if (t < 128) redB[t] = sArr[t];
    __syncthreads();
    for (int off = 64; off > 0; off >>= 1) {
        if (t < off) redB[t] = fmaxf(redB[t], redB[t + off]);
        __syncthreads();
    }
    const float m_c = redB[0];
    __syncthreads();
    if (t < 128) wArr[t] = __expf(sArr[t] - m_c);
    __syncthreads();
    if (t < 128) redB[t] = wArr[t];
    __syncthreads();
    for (int off = 64; off > 0; off >>= 1) {
        if (t < off) redB[t] += redB[t + off];
        __syncthreads();
    }

    // ---- partial context from the LDS tile (no extra HBM read)
    const int pbase = (b * NCHUNK_ + chunk) * H_;
    float a0 = 0.f, a1 = 0.f;
    for (int r = 0; r < 128; ++r) {
        const float w = wArr[r];
        a0 += w * bf2f(As[r * KP_ + t]);
        a1 += w * bf2f(As[r * KP_ + t + 256]);
    }
    partCtx[pbase + t]       = a0;
    partCtx[pbase + t + 256] = a1;
    if (t == 0) {
        partM[b * NCHUNK_ + chunk] = m_c;
        partL[b * NCHUNK_ + chunk] = redB[0];
    }
}

// ---------------------------------------------------------------------------
// Combine chunk partials: global softmax rescale -> context + normalized attn
// ---------------------------------------------------------------------------
__global__ void k_combine(const float* __restrict__ partM, const float* __restrict__ partL,
                          const float* __restrict__ partCtx,
                          float* __restrict__ attn, float* __restrict__ context) {
    __shared__ float sM, sL;
    __shared__ float coef[NCHUNK_];
    const int b = blockIdx.x;
    const int t = threadIdx.x;
    if (t == 0) {
        float gm = -1e30f;
        for (int c = 0; c < NCHUNK_; ++c) gm = fmaxf(gm, partM[b * NCHUNK_ + c]);
        float L = 0.f;
        for (int c = 0; c < NCHUNK_; ++c) {
            const float cf = __expf(partM[b * NCHUNK_ + c] - gm);
            coef[c] = cf;
            L += partL[b * NCHUNK_ + c] * cf;
        }
        sM = gm; sL = L;
    }
    __syncthreads();
    const float gm   = sM;
    const float invL = 1.f / sL;

    for (int h = t; h < H_; h += 256) {
        float a = 0.f;
#pragma unroll
        for (int c = 0; c < NCHUNK_; ++c)
            a += partCtx[(b * NCHUNK_ + c) * H_ + h] * coef[c];
        context[b * H_ + h] = a * invL;
    }
    float* row = attn + (long)b * S_;
    for (int s = t; s < S_; s += 256) row[s] = __expf(row[s] - gm) * invL;
}

// ---------------------------------------------------------------------------
// GRU r/z gates as WMMA GEMMs: M=64 (4 M-tiles per wave), N=512, K=768+512.
// Each weight fragment is loaded once and reused across all 4 M-tiles, so the
// gate weights touch memory ~once.  Epilogue emits zg and rh = sigmoid(r)*hidden.
// ---------------------------------------------------------------------------
__global__ void k_gates_rz(const float* __restrict__ x, const float* __restrict__ context,
                           const float* __restrict__ hidden,
                           const float* __restrict__ Wxr, const float* __restrict__ bxr,
                           const float* __restrict__ Whr,
                           const float* __restrict__ Wxz, const float* __restrict__ bxz,
                           const float* __restrict__ Whz,
                           float* __restrict__ rh, float* __restrict__ zg) {
    const int lane = threadIdx.x & 31;
    const int wave = threadIdx.x >> 5;
    const int n0   = (blockIdx.x * 8 + wave) * 16;

    v8f zero = {};
    v8f ra[4] = {zero, zero, zero, zero};
    v8f za[4] = {zero, zero, zero, zero};

    for (int k0 = 0; k0 < IN_ + H_; k0 += 32) {        // xc @ Wx{r,z}
        v16bf br = load_b_frag_f32(Wxr, H_, k0, n0, lane);
        v16bf bz = load_b_frag_f32(Wxz, H_, k0, n0, lane);
#pragma unroll
        for (int m = 0; m < 4; ++m) {
            v16bf a = load_a_xc(x, context, m * 16, k0, lane);
            ra[m] = __builtin_amdgcn_wmma_f32_16x16x32_bf16(
                false, a, false, br, (short)0, ra[m], false, false);
            za[m] = __builtin_amdgcn_wmma_f32_16x16x32_bf16(
                false, a, false, bz, (short)0, za[m], false, false);
        }
    }
    for (int k0 = 0; k0 < H_; k0 += 32) {              // hidden @ Wh{r,z}
        v16bf br = load_b_frag_f32(Whr, H_, k0, n0, lane);
        v16bf bz = load_b_frag_f32(Whz, H_, k0, n0, lane);
#pragma unroll
        for (int m = 0; m < 4; ++m) {
            v16bf a = load_a_frag_f32(hidden, H_, m * 16, k0, lane);
            ra[m] = __builtin_amdgcn_wmma_f32_16x16x32_bf16(
                false, a, false, br, (short)0, ra[m], false, false);
            za[m] = __builtin_amdgcn_wmma_f32_16x16x32_bf16(
                false, a, false, bz, (short)0, za[m], false, false);
        }
    }
    const int   col = n0 + (lane & 15);
    const float brb = bxr[col];
    const float bzb = bxz[col];
#pragma unroll
    for (int m = 0; m < 4; ++m) {
#pragma unroll
        for (int i = 0; i < 8; ++i) {
            const int row = m * 16 + i + ((lane >> 4) << 3);   // batch index
            const float r = fast_sigmoid(ra[m][i] + brb);
            const float z = fast_sigmoid(za[m][i] + bzb);
            rh[row * H_ + col] = r * hidden[row * H_ + col];
            zg[row * H_ + col] = z;
        }
    }
}

// h_tilde = tanh(xc @ Wxh + b + rh @ Whh); new_hidden = z*hidden + (1-z)*h_tilde
__global__ void k_gates_h(const float* __restrict__ x, const float* __restrict__ context,
                          const float* __restrict__ hidden,
                          const float* __restrict__ Wxh, const float* __restrict__ bxh,
                          const float* __restrict__ Whh,
                          const float* __restrict__ rh, const float* __restrict__ zg,
                          float* __restrict__ new_hidden) {
    const int lane = threadIdx.x & 31;
    const int wave = threadIdx.x >> 5;
    const int n0   = (blockIdx.x * 8 + wave) * 16;

    v8f zero = {};
    v8f acc[4] = {zero, zero, zero, zero};

    for (int k0 = 0; k0 < IN_ + H_; k0 += 32) {        // xc @ Wxh
        v16bf bb = load_b_frag_f32(Wxh, H_, k0, n0, lane);
#pragma unroll
        for (int m = 0; m < 4; ++m) {
            v16bf a = load_a_xc(x, context, m * 16, k0, lane);
            acc[m] = __builtin_amdgcn_wmma_f32_16x16x32_bf16(
                false, a, false, bb, (short)0, acc[m], false, false);
        }
    }
    for (int k0 = 0; k0 < H_; k0 += 32) {              // (r*hidden) @ Whh
        v16bf bb = load_b_frag_f32(Whh, H_, k0, n0, lane);
#pragma unroll
        for (int m = 0; m < 4; ++m) {
            v16bf a = load_a_frag_f32(rh, H_, m * 16, k0, lane);
            acc[m] = __builtin_amdgcn_wmma_f32_16x16x32_bf16(
                false, a, false, bb, (short)0, acc[m], false, false);
        }
    }
    const int   col = n0 + (lane & 15);
    const float bb  = bxh[col];
#pragma unroll
    for (int m = 0; m < 4; ++m) {
#pragma unroll
        for (int i = 0; i < 8; ++i) {
            const int row = m * 16 + i + ((lane >> 4) << 3);
            const float ht = fast_tanh(acc[m][i] + bb);
            const float z  = zg[row * H_ + col];
            new_hidden[row * H_ + col] =
                z * hidden[row * H_ + col] + (1.f - z) * ht;
        }
    }
}

// ---------------------------------------------------------------------------
// logits = new_hidden @ Who + b.  One wave per 16-col N-tile; each Who
// B-fragment (f32->bf16 in registers) is reused across all 4 M-tiles, and each
// lane prefetches a distinct row of the fragment two k-steps ahead
// (global_prefetch_b8) to keep the 65 MB Who stream at line rate.
// ---------------------------------------------------------------------------
__global__ void k_output(const float* __restrict__ nh, const float* __restrict__ Who,
                         const float* __restrict__ bo, float* __restrict__ out) {
    const int lane = threadIdx.x & 31;
    const int wave = threadIdx.x >> 5;
    const int n0   = (blockIdx.x * 8 + wave) * 16;

    v8f zero = {};
    v8f acc[4] = {zero, zero, zero, zero};

    for (int k0 = 0; k0 < H_; k0 += 32) {
        if (k0 + 64 < H_)
            __builtin_prefetch(Who + (long)(k0 + 64 + lane) * OUT_ + n0, 0, 0);
        v16bf bb = load_b_frag_f32(Who, OUT_, k0, n0, lane);
#pragma unroll
        for (int m = 0; m < 4; ++m) {
            v16bf a = load_a_frag_f32(nh, H_, m * 16, k0, lane);
            acc[m] = __builtin_amdgcn_wmma_f32_16x16x32_bf16(
                false, a, false, bb, (short)0, acc[m], false, false);
        }
    }
    const int   col  = n0 + (lane & 15);
    const float bias = bo[col];
#pragma unroll
    for (int m = 0; m < 4; ++m) {
#pragma unroll
        for (int i = 0; i < 8; ++i) {
            const int row = m * 16 + i + ((lane >> 4) << 3);
            out[(long)row * OUT_ + col] = acc[m][i] + bias;
        }
    }
}

// ---------------------------------------------------------------------------
extern "C" void kernel_launch(void* const* d_in, const int* in_sizes, int n_in,
                              void* d_out, int out_size, void* d_ws, size_t ws_size,
                              hipStream_t stream) {
    (void)in_sizes; (void)n_in; (void)out_size; (void)ws_size;
    const float* x      = (const float*)d_in[0];
    const float* hidden = (const float*)d_in[1];
    const float* enc    = (const float*)d_in[2];
    const float* W1_w   = (const float*)d_in[3];
    const float* W1_b   = (const float*)d_in[4];
    const float* W2_w   = (const float*)d_in[5];
    const float* v_w    = (const float*)d_in[6];
    const float* Wxr_w  = (const float*)d_in[7];
    const float* Wxr_b  = (const float*)d_in[8];
    const float* Whr_w  = (const float*)d_in[9];
    const float* Wxz_w  = (const float*)d_in[10];
    const float* Wxz_b  = (const float*)d_in[11];
    const float* Whz_w  = (const float*)d_in[12];
    const float* Wxh_w  = (const float*)d_in[13];
    const float* Wxh_b  = (const float*)d_in[14];
    const float* Whh_w  = (const float*)d_in[15];
    const float* Who_w  = (const float*)d_in[16];
    const float* Who_b  = (const float*)d_in[17];

    float* out_logits = (float*)d_out;                     // [B, OUT]
    float* out_newh   = out_logits + (long)B_ * OUT_;      // [B, H]
    float* out_attn   = out_newh + (long)B_ * H_;          // [B, S] (raw -> normalized)

    float* ws      = (float*)d_ws;
    float* pre     = ws;                               // B*E          = 16384
    float* context = pre + B_ * E_;                    // B*H          = 32768
    float* rh      = context + B_ * H_;                // B*H  (r * hidden)
    float* zg      = rh + B_ * H_;                     // B*H
    float* partM   = zg + B_ * H_;                     // B*NCHUNK     = 1024
    float* partL   = partM + B_ * NCHUNK_;             // B*NCHUNK
    float* partCtx = partL + B_ * NCHUNK_;             // B*NCHUNK*H   = 524288
    __bf16* W1p    = (__bf16*)(partCtx + (long)B_ * NCHUNK_ * H_);  // H*E bf16

    const size_t ldsAttn = (size_t)128 * KP_ * sizeof(__bf16) + 384 * sizeof(float);

    k_pre<<<dim3((B_ * E_) / 256), 256, 0, stream>>>(hidden, W2_w, W1_b, pre);
    k_pack_w1<<<dim3((H_ * E_) / 256), 256, 0, stream>>>(W1_w, W1p);
    k_attn_chunk<<<dim3(NCHUNK_, B_), 256, ldsAttn, stream>>>(
        enc, W1p, pre, v_w, out_attn, partM, partL, partCtx);
    k_combine<<<dim3(B_), 256, 0, stream>>>(partM, partL, partCtx, out_attn, context);
    k_gates_rz<<<dim3((H_ / 16) / 8), 256, 0, stream>>>(x, context, hidden,
        Wxr_w, Wxr_b, Whr_w, Wxz_w, Wxz_b, Whz_w, rh, zg);
    k_gates_h<<<dim3((H_ / 16) / 8), 256, 0, stream>>>(x, context, hidden,
        Wxh_w, Wxh_b, Whh_w, rh, zg, out_newh);
    k_output<<<dim3(OUT_ / 128), 256, 0, stream>>>(out_newh, Who_w, Who_b, out_logits);
}
